// HypergraphEmbeddingAggregator_47278999994504
// MI455X (gfx1250) — matrix operand order repaired and checked by reading, not compiled, not run on hardware
//
#include <hip/hip_runtime.h>
#include <hip/hip_bf16.h>

// ---------------------------------------------------------------------------
// HypergraphEmbeddingAggregator for MI455X (gfx1250, wave32, WMMA)
//   phase 1: visit = H^T @ X          (split-K bf16 WMMA + f32 atomics)
//   phase 2: gx    = visit @ W_ih^T+b (bf16 WMMA)
//   phase 2b: W_hh -> float4-blocked transpose (coalesced GRU streaming)
//   phase 3: sequential GRU scan (1 WGP, h in LDS, L2-resident weights)
//   phase 4: attention softmax + weighted sum -> out[256]
// NOTE: 32 | 100000, so every 32-deep K tile is full -> no guarded tail path.
// ---------------------------------------------------------------------------

typedef __attribute__((ext_vector_type(16))) __bf16 v16bf;
typedef __attribute__((ext_vector_type(8)))  __bf16 v8bf;
typedef __attribute__((ext_vector_type(4)))  __bf16 v4bf;
typedef __attribute__((ext_vector_type(8)))  float  v8f;

#define NN     100000
#define TT     512
#define DIN    256
#define DH     256
#define G3     768      // 3*DH
#define TILEK  32
#define KCHUNK 2048
#define LROW   40       // LDS row stride in bf16 (80B: 16B-aligned, 20-word banking)

// ---------------- zero workspace (atomic accumulation target) --------------
__global__ void k_zero(float4* __restrict__ p) {
    p[blockIdx.x * 256 + threadIdx.x] = make_float4(0.f, 0.f, 0.f, 0.f);
}

// build v16bf fragment from two contiguous 16B LDS runs -> 2x ds_load_b128
__device__ __forceinline__ v16bf frag16(const __bf16* lo, const __bf16* hi) {
    v8bf a0 = *(const v8bf*)lo;
    v8bf a1 = *(const v8bf*)hi;
    return __builtin_shufflevector(a0, a1, 0, 1, 2, 3, 4, 5, 6, 7,
                                           8, 9, 10, 11, 12, 13, 14, 15);
}

// ---------------- phase 1: visit = H^T @ X (split-K WMMA) -------------------
// grid: (16 t-blocks of 32 rows, 49 k-chunks), block: 512 threads = 16 waves
// wave w owns output columns d in [16w,16w+16); MT=2 t-tiles share one B frag
__global__ __launch_bounds__(512)
void k_visit_gemm(const float* __restrict__ H, const float* __restrict__ X,
                  float* __restrict__ visit) {
    __shared__ __align__(16) __bf16 a_s[2][16][LROW];  // [mt][m][k]   A = H^T
    __shared__ __align__(16) __bf16 b_s[DIN][LROW];    // [n][k] n-major  B = X

    const int tid  = threadIdx.x;
    const int wave = tid >> 5;
    const int lane = tid & 31;
    const int half = lane >> 4;
    const int lr   = lane & 15;
    const int t0   = blockIdx.x * 32;
    const int kb0  = blockIdx.y * KCHUNK;
    const int d0   = wave * 16;
    // uniform, exact trip count: 32 | 100000 so every tile is full
    const int rem = NN - kb0;
    const int kt  = ((KCHUNK < rem) ? KCHUNK : rem) / TILEK;

    v8f acc0, acc1;
#pragma unroll
    for (int i = 0; i < 8; ++i) { acc0[i] = 0.f; acc1[i] = 0.f; }

    for (int it = 0; it < kt; ++it) {
        const int kb = kb0 + it * TILEK;
        __syncthreads();
        // A: 1024 elems (32 t-rows x 32 k), 2/thread, coalesced over t
        float av[2];
#pragma unroll
        for (int r = 0; r < 2; ++r) {
            const int idx = r * 512 + tid;
            const int k = idx >> 5, m2 = idx & 31;
            av[r] = H[(long)(kb + k) * TT + t0 + m2];
        }
        // B: 8192 elems, 4x float4/thread, coalesced over n
        float4 xv[4];
#pragma unroll
        for (int r = 0; r < 4; ++r) {
            const int idx4 = r * 512 + tid;
            const int k = idx4 >> 6, n4 = idx4 & 63;
            xv[r] = *(const float4*)&X[(long)(kb + k) * DIN + 4 * n4];
        }
        // speculative prefetch of next tile (OOB translation silently dropped)
        __builtin_prefetch(&H[(long)(kb + TILEK + (tid >> 5)) * TT + t0 + (tid & 31)], 0, 1);
        __builtin_prefetch(&X[(long)(kb + TILEK + (tid >> 6)) * DIN + 4 * (tid & 63)], 0, 1);
        // convert + LDS stores (after all loads issued)
#pragma unroll
        for (int r = 0; r < 2; ++r) {
            const int idx = r * 512 + tid;
            const int k = idx >> 5, m2 = idx & 31;
            a_s[m2 >> 4][m2 & 15][k] = (__bf16)av[r];
        }
#pragma unroll
        for (int r = 0; r < 4; ++r) {
            const int idx4 = r * 512 + tid;
            const int k = idx4 >> 6, n4 = idx4 & 63;
            b_s[4 * n4 + 0][k] = (__bf16)xv[r].x;
            b_s[4 * n4 + 1][k] = (__bf16)xv[r].y;
            b_s[4 * n4 + 2][k] = (__bf16)xv[r].z;
            b_s[4 * n4 + 3][k] = (__bf16)xv[r].w;
        }
        __syncthreads();

        // fragments: contiguous 16B runs -> ds_load_b128 pairs
        const v16bf bf = frag16(&b_s[d0 + lr][half * 8],
                                &b_s[d0 + lr][16 + half * 8]);
        const v16bf af0 = frag16(&a_s[0][lr][half * 8],
                                 &a_s[0][lr][16 + half * 8]);
        const v16bf af1 = frag16(&a_s[1][lr][half * 8],
                                 &a_s[1][lr][16 + half * 8]);
        acc0 = __builtin_amdgcn_wmma_f32_16x16x32_bf16(
                   false, af0, false, bf, (short)0, acc0, false, false);
        acc1 = __builtin_amdgcn_wmma_f32_16x16x32_bf16(
                   false, af1, false, bf, (short)0, acc1, false, false);
    }

    // C layout: VGPR v -> M = v + 8*half, N = lr; split-K f32 atomics
#pragma unroll
    for (int v = 0; v < 8; ++v) {
        const int m = v + 8 * half;
        atomicAdd(&visit[(t0 + m) * DIN + d0 + lr], acc0[v]);
        atomicAdd(&visit[(t0 + 16 + m) * DIN + d0 + lr], acc1[v]);
    }
}

// ---------------- phase 2: gx = visit @ W_ih^T + b_ih -----------------------
// grid: (16 t-blocks, 3 n-blocks), block 512 = 16 waves; MT=2 t-tiles
__global__ __launch_bounds__(512)
void k_gx_gemm(const float* __restrict__ visit, const float* __restrict__ W_ih,
               const float* __restrict__ b_ih, float* __restrict__ gx) {
    __shared__ __align__(16) __bf16 a_s[2][16][LROW];  // [mt][m][k] visit tile
    __shared__ __align__(16) __bf16 b_s[DIN][LROW];    // [n][k]     W_ih tile

    const int tid  = threadIdx.x;
    const int wave = tid >> 5;
    const int lane = tid & 31;
    const int half = lane >> 4;
    const int lr   = lane & 15;
    const int t0   = blockIdx.x * 32;
    const int n0   = blockIdx.y * 256;
    const int d0   = wave * 16;

    v8f acc0, acc1;
#pragma unroll
    for (int i = 0; i < 8; ++i) { acc0[i] = 0.f; acc1[i] = 0.f; }

    for (int kb = 0; kb < DIN; kb += TILEK) {
        __syncthreads();
        {   // A: 1024 elems, 2/thread, coalesced over k within a visit row
            float av[2];
#pragma unroll
            for (int r = 0; r < 2; ++r) {
                const int idx = r * 512 + tid;
                const int m2 = idx >> 5, k = idx & 31;
                av[r] = visit[(t0 + m2) * DIN + kb + k];
            }
#pragma unroll
            for (int r = 0; r < 2; ++r) {
                const int idx = r * 512 + tid;
                const int m2 = idx >> 5, k = idx & 31;
                a_s[m2 >> 4][m2 & 15][k] = (__bf16)av[r];
            }
        }
        {   // B: W_ih rows are contiguous in k -> float4 load + v4bf store
            float4 wv[4];
#pragma unroll
            for (int r = 0; r < 4; ++r) {
                const int idx4 = r * 512 + tid;
                const int n = idx4 >> 3, k4 = idx4 & 7;
                wv[r] = *(const float4*)&W_ih[(n0 + n) * DIN + kb + 4 * k4];
            }
#pragma unroll
            for (int r = 0; r < 4; ++r) {
                const int idx4 = r * 512 + tid;
                const int n = idx4 >> 3, k4 = idx4 & 7;
                v4bf pk;
                pk[0] = (__bf16)wv[r].x; pk[1] = (__bf16)wv[r].y;
                pk[2] = (__bf16)wv[r].z; pk[3] = (__bf16)wv[r].w;
                *(v4bf*)&b_s[n][4 * k4] = pk;
            }
        }
        __syncthreads();

        const v16bf bf = frag16(&b_s[d0 + lr][half * 8],
                                &b_s[d0 + lr][16 + half * 8]);
        const v16bf af0 = frag16(&a_s[0][lr][half * 8],
                                 &a_s[0][lr][16 + half * 8]);
        const v16bf af1 = frag16(&a_s[1][lr][half * 8],
                                 &a_s[1][lr][16 + half * 8]);
        acc0 = __builtin_amdgcn_wmma_f32_16x16x32_bf16(
                   false, af0, false, bf, (short)0, acc0, false, false);
        acc1 = __builtin_amdgcn_wmma_f32_16x16x32_bf16(
                   false, af1, false, bf, (short)0, acc1, false, false);
    }

    const int ncol = n0 + d0 + lr;
    const float bias = b_ih[ncol];
#pragma unroll
    for (int v = 0; v < 8; ++v) {
        const int m = v + 8 * half;
        gx[(t0 + m) * G3 + ncol]      = acc0[v] + bias;
        gx[(t0 + 16 + m) * G3 + ncol] = acc1[v] + bias;
    }
}

// ------------- phase 2b: W_hh -> float4-blocked transpose -------------------
// out[(k/4)*768 + j] = {W_hh[j][k..k+3]}  => GRU streams coalesced b128 loads
__global__ void k_whh_tr(const float* __restrict__ W_hh, float4* __restrict__ o) {
    const int idx = blockIdx.x * 256 + threadIdx.x;   // 192*256 = 49152
    const int k4 = idx / G3, j = idx % G3;
    const float* r = &W_hh[j * DH + 4 * k4];
    o[idx] = make_float4(r[0], r[1], r[2], r[3]);
}

// ---------------- phase 3: sequential GRU scan (single WGP) -----------------
__global__ __launch_bounds__(768)
void k_gru(const float4* __restrict__ whhT4, const float* __restrict__ gx,
           const float* __restrict__ b_hh, float* __restrict__ hidden) {
    __shared__ __align__(16) float h_s[DH];
    __shared__ float gh_s[G3];
    const int j = threadIdx.x;
    const float bias = b_hh[j];
    if (j < DH) h_s[j] = 0.f;
    __syncthreads();

    for (int t = 0; t < TT; ++t) {
        float acc = bias;
#pragma unroll 8
        for (int k4 = 0; k4 < DH / 4; ++k4) {      // coalesced b128 weight loads
            const float4 w = whhT4[k4 * G3 + j];
            const float4 h = *(const float4*)&h_s[4 * k4];   // ds_load_b128
            acc += w.x * h.x + w.y * h.y + w.z * h.z + w.w * h.w;
        }
        gh_s[j] = acc;
        __syncthreads();
        if (j < DH) {                               // torch gate order r,z,n
            const float* gxt = gx + t * G3;
            const float r = 1.f / (1.f + __expf(-(gxt[j] + gh_s[j])));
            const float z = 1.f / (1.f + __expf(-(gxt[DH + j] + gh_s[DH + j])));
            const float n = tanhf(gxt[2 * DH + j] + r * gh_s[2 * DH + j]);
            const float hn = (1.f - z) * n + z * h_s[j];
            hidden[t * DH + j] = hn;
            h_s[j] = hn;
        }
        __syncthreads();
    }
}

// ---------------- phase 4: softmax attention + weighted sum -----------------
__global__ __launch_bounds__(512)
void k_att(const float* __restrict__ hidden, const float* __restrict__ w_att,
           float* __restrict__ out) {
    __shared__ __align__(16) float w_s[DH];
    __shared__ float red[TT];
    __shared__ float alpha[TT];
    const int t = threadIdx.x;
    if (t < DH) w_s[t] = w_att[t];
    __syncthreads();

    float logit = 0.f;
#pragma unroll 4
    for (int k4 = 0; k4 < DH / 4; ++k4) {
        const float4 h = *(const float4*)&hidden[t * DH + 4 * k4];
        const float4 w = *(const float4*)&w_s[4 * k4];
        logit += h.x * w.x + h.y * w.y + h.z * w.z + h.w * w.w;
    }
    red[t] = logit;
    __syncthreads();
    for (int s = TT / 2; s > 0; s >>= 1) {
        if (t < s) red[t] = fmaxf(red[t], red[t + s]);
        __syncthreads();
    }
    const float mx = red[0];
    __syncthreads();
    const float e = __expf(logit - mx);
    alpha[t] = e;
    red[t] = e;
    __syncthreads();
    for (int s = TT / 2; s > 0; s >>= 1) {
        if (t < s) red[t] += red[t + s];
        __syncthreads();
    }
    const float inv = 1.f / red[0];
    __syncthreads();
    if (t < DH) {
        float o = 0.f;
#pragma unroll 4
        for (int tt = 0; tt < TT; ++tt) o += alpha[tt] * hidden[tt * DH + t];
        out[t] = o * inv;
    }
}

// ---------------------------------------------------------------------------
extern "C" void kernel_launch(void* const* d_in, const int* in_sizes, int n_in,
                              void* d_out, int out_size, void* d_ws, size_t ws_size,
                              hipStream_t stream) {
    const float* X     = (const float*)d_in[0];
    const float* H     = (const float*)d_in[1];
    const float* W_ih  = (const float*)d_in[2];
    const float* W_hh  = (const float*)d_in[3];
    const float* b_ih  = (const float*)d_in[4];
    const float* b_hh  = (const float*)d_in[5];
    const float* w_att = (const float*)d_in[6];
    float* out = (float*)d_out;

    float* ws     = (float*)d_ws;
    float* visit  = ws;                       // 512*256        = 131072 f32
    float* gx     = ws + 131072;              // 512*768        = 393216 f32
    float* hidden = ws + 524288;              // 512*256        = 131072 f32
    float4* whhT4 = (float4*)(ws + 655360);   // 64*768 float4  = 196608 f32

    // phase 0: zero split-K accumulator (131072 f32 = 32768 float4)
    k_zero<<<128, 256, 0, stream>>>((float4*)visit);
    // phase 1: visit = H^T @ X   (49 K-chunks of 2048, covers 100000)
    k_visit_gemm<<<dim3(16, 49), 512, 0, stream>>>(H, X, visit);
    // phase 2: gx = visit @ W_ih^T + b_ih
    k_gx_gemm<<<dim3(16, 3), 512, 0, stream>>>(visit, W_ih, b_ih, gx);
    // phase 2b: transpose W_hh for coalesced GRU streaming
    k_whh_tr<<<192, 256, 0, stream>>>(W_hh, whhT4);
    // phase 3: sequential GRU scan on one WGP
    k_gru<<<1, 768, 0, stream>>>(whhT4, gx, b_hh, hidden);
    // phase 4: attention-weighted sum -> out[256]
    k_att<<<1, 512, 0, stream>>>(hidden, w_att, out);
}
